// GSSupervised_53386443489817
// MI455X (gfx1250) — compile-verified, matrix operand order checked by reading, not run on hardware
//
#include <hip/hip_runtime.h>

// ---------------------------------------------------------------------------
// GraphSAGE supervised forward on gfx1250 (MI455X), bf16 WMMA w/ f32 accum.
//
// Matrix path: v_wmma_f32_16x16x32_bf16 (probe-confirmed builtin).
// Assumed fragment layouts (per CDNA5 ISA 7.12.2, wave32):
//   A (16x32 bf16): lane m(0..15) holds row m, K = {0..7,16..23}; lane m+16
//                   holds row m, K = {8..15,24..31}; 16 bf16 per lane in
//                   K-ascending order within each 8-group.
//   B (32x16 bf16): mirrored -- lane n holds column n with the same K split.
//                   Weights are PRE-PACKED into this per-lane order so the
//                   GEMM inner loop is two 16B loads per fragment.
//   C/D (16x16 f32): VGPR j, lanes 0-15 -> (M=j, N=lane);
//                    lanes 16-31 -> (M=j+8, N=lane-16).
// ---------------------------------------------------------------------------

typedef __attribute__((ext_vector_type(8)))  __bf16 v8bf;
typedef __attribute__((ext_vector_type(16))) __bf16 v16bf;
typedef __attribute__((ext_vector_type(8)))  float  v8f;

#define DIN    256   // feature dim (== 2*D_HID, so layer1 and layer2 share shapes)
#define DHID   128
#define NCLS   40
#define KTILES 8     // 256 / 32
#define PSTR   264   // LDS panel row stride in bf16 elems (256 + 8 pad: conflict-free)

// ---- fp32 -> bf16 (round to nearest even) ----------------------------------
__device__ __forceinline__ unsigned short f2bf(float f) {
  unsigned int u = __builtin_bit_cast(unsigned int, f);
  u += 0x7FFFu + ((u >> 16) & 1u);
  return (unsigned short)(u >> 16);
}

__device__ __forceinline__ v16bf frag_from(const unsigned short* p0,
                                           const unsigned short* p1) {
  v8bf lo = *(const v8bf*)p0;
  v8bf hi = *(const v8bf*)p1;
  return __builtin_shufflevector(lo, hi, 0,1,2,3,4,5,6,7,8,9,10,11,12,13,14,15);
}

__device__ __forceinline__ void acc4(float4& a, float4 v) {
  a.x += v.x; a.y += v.y; a.z += v.z; a.w += v.w;
}

// ---- weight pre-pack: W[K=256][N] (fp32, two halves) -> bf16 B-fragments ---
// Packed element index: ((nt*KTILES + kt)*512) + lane*16 + j
__global__ void pack2_kernel(const float* __restrict__ wa,
                             const float* __restrict__ wb,
                             unsigned short* __restrict__ outp, int total) {
  int idx = blockIdx.x * 256 + threadIdx.x;
  if (idx >= total) return;
  int blk = idx >> 9;
  int l   = (idx >> 4) & 31;
  int j   = idx & 15;
  int nt = blk >> 3, kt = blk & 7;
  int base8 = (l & 16) ? 8 : 0;
  int kk = (j < 8) ? (base8 + j) : (base8 + 8 + j);   // {0..7,16..23} / {8..15,24..31}
  int K = kt * 32 + kk;
  int n = nt * 16 + (l & 15);
  float v = (n < DHID) ? wa[K * DHID + n] : wb[K * DHID + (n - DHID)];
  outp[idx] = f2bf(v);
}

__global__ void pack1_kernel(const float* __restrict__ w,
                             unsigned short* __restrict__ outp, int N, int total) {
  int idx = blockIdx.x * 256 + threadIdx.x;
  if (idx >= total) return;
  int blk = idx >> 9;
  int l   = (idx >> 4) & 31;
  int j   = idx & 15;
  int nt = blk >> 3, kt = blk & 7;
  int base8 = (l & 16) ? 8 : 0;
  int kk = (j < 8) ? (base8 + j) : (base8 + 8 + j);
  int K = kt * 32 + kk;
  int n = nt * 16 + (l & 15);
  float v = (n < N) ? w[K * N + n] : 0.f;                // zero-pad cols 40..47
  outp[idx] = f2bf(v);
}

// ---- fused gather + mean-agg + dual GEMM + ReLU ----------------------------
// out[row, 0:128]   = relu( x[row]   @ Wx )    (Wpack ntiles 0..7)
// out[row, 128:256] = relu( mean_S(neib) @ Wn )(Wpack ntiles 8..15)
__global__ __launch_bounds__(256)
void sage_layer_kernel(const float* __restrict__ xsrc, const int* __restrict__ xids,
                       const float* __restrict__ nsrc, const int* __restrict__ nids,
                       int S, float invS,
                       const unsigned short* __restrict__ Wpack,
                       float* __restrict__ out) {
  __shared__ unsigned short Xp[16 * PSTR];   // self panel   (bf16)
  __shared__ unsigned short Ap[16 * PSTR];   // neighbor-mean panel (bf16)

  const int tid = threadIdx.x;
  const int r   = tid >> 4;        // 0..15 : local row
  const int cq  = tid & 15;        // 16 threads per row, float4 granularity
  const int row = blockIdx.x * 16 + r;

  // Phase 1a: gather self row -> bf16 panel (coalesced float4)
  {
    int sid = xids ? xids[row] : row;
    const float4* src = (const float4*)(xsrc + (size_t)sid * DIN);
    #pragma unroll
    for (int c = 0; c < 4; ++c) {
      float4 v = src[cq + c * 16];
      unsigned short* d = &Xp[r * PSTR + (cq + c * 16) * 4];
      d[0] = f2bf(v.x); d[1] = f2bf(v.y); d[2] = f2bf(v.z); d[3] = f2bf(v.w);
    }
  }
  // Phase 1b: mean over S sampled neighbors -> bf16 panel
  {
    float4 a0 = {0,0,0,0}, a1 = {0,0,0,0}, a2 = {0,0,0,0}, a3 = {0,0,0,0};
    for (int j = 0; j < S; ++j) {
      int nid = nids ? nids[row * S + j] : (row * S + j);
      const float4* src = (const float4*)(nsrc + (size_t)nid * DIN);
      acc4(a0, src[cq]); acc4(a1, src[cq + 16]);
      acc4(a2, src[cq + 32]); acc4(a3, src[cq + 48]);
    }
    float4 m[4] = {a0, a1, a2, a3};
    #pragma unroll
    for (int c = 0; c < 4; ++c) {
      unsigned short* d = &Ap[r * PSTR + (cq + c * 16) * 4];
      d[0] = f2bf(m[c].x * invS); d[1] = f2bf(m[c].y * invS);
      d[2] = f2bf(m[c].z * invS); d[3] = f2bf(m[c].w * invS);
    }
  }
  __syncthreads();

  // Phase 2: each wave -> two 16x16 output tiles; K-loop of 8 bf16 WMMAs each
  const int wave  = tid >> 5;              // 0..7
  const int lane  = tid & 31;
  const int m     = lane & 15;
  const int base8 = (lane & 16) ? 8 : 0;
  const unsigned short* panel = (wave < 4) ? Xp : Ap;
  const int n0 = 2 * wave;                 // ntile ids 0..15 (0..7 = X, 8..15 = Agg)

  v8f c0 = {}; v8f c1 = {};
  #pragma unroll
  for (int kt = 0; kt < KTILES; ++kt) {
    const unsigned short* ap = &panel[m * PSTR + kt * 32 + base8];
    v16bf A = frag_from(ap, ap + 16);
    const unsigned short* wp0 = Wpack + (((n0 * KTILES + kt) << 9) + lane * 16);
    const unsigned short* wp1 = wp0 + (KTILES << 9);      // next ntile
    v16bf B0 = frag_from(wp0, wp0 + 8);
    v16bf B1 = frag_from(wp1, wp1 + 8);
    c0 = __builtin_amdgcn_wmma_f32_16x16x32_bf16(false, A, false, B0, (short)0, c0, false, false);
    c1 = __builtin_amdgcn_wmma_f32_16x16x32_bf16(false, A, false, B1, (short)0, c1, false, false);
  }

  const int outBase = blockIdx.x * 16;
  #pragma unroll
  for (int j = 0; j < 8; ++j) {
    int ro = outBase + j + base8;
    out[(size_t)ro * DIN + n0 * 16 + m]       = fmaxf(c0[j], 0.f);
    out[(size_t)ro * DIN + (n0 + 1) * 16 + m] = fmaxf(c1[j], 0.f);
  }
}

// ---- L2-normalize rows + final classifier (WMMA against padded w_fc) -------
__global__ __launch_bounds__(128)
void final_kernel(const float* __restrict__ b0,
                  const unsigned short* __restrict__ WFp,
                  const float* __restrict__ bfc,
                  float* __restrict__ out) {
  __shared__ unsigned short P[16 * PSTR];
  __shared__ float partial[128];
  __shared__ float invn[16];

  const int tid = threadIdx.x;
  const int r   = tid >> 3;       // 0..15
  const int c8  = tid & 7;        // 8 threads per row, each 8 float4 chunks
  const int row = blockIdx.x * 16 + r;

  float4 v[8];
  const float4* src = (const float4*)(b0 + (size_t)row * DIN);
  float ss = 0.f;
  #pragma unroll
  for (int i = 0; i < 8; ++i) {
    v[i] = src[c8 + i * 8];
    ss += v[i].x * v[i].x + v[i].y * v[i].y + v[i].z * v[i].z + v[i].w * v[i].w;
  }
  partial[tid] = ss;
  __syncthreads();
  if (tid < 16) {
    float s = 0.f;
    #pragma unroll
    for (int i = 0; i < 8; ++i) s += partial[tid * 8 + i];
    invn[tid] = 1.f / fmaxf(sqrtf(s), 1e-12f);
  }
  __syncthreads();
  const float sc = invn[r];
  #pragma unroll
  for (int i = 0; i < 8; ++i) {
    unsigned short* d = &P[r * PSTR + (c8 + i * 8) * 4];
    d[0] = f2bf(v[i].x * sc); d[1] = f2bf(v[i].y * sc);
    d[2] = f2bf(v[i].z * sc); d[3] = f2bf(v[i].w * sc);
  }
  __syncthreads();

  const int wave = tid >> 5;      // 0..3; waves 0..2 compute 3 ntiles (48 padded cols)
  const int lane = tid & 31;
  if (wave < 3) {                 // wave-uniform branch: EXEC all-ones inside
    const int m = lane & 15, base8 = (lane & 16) ? 8 : 0;
    v8f c = {};
    #pragma unroll
    for (int kt = 0; kt < KTILES; ++kt) {
      const unsigned short* ap = &P[m * PSTR + kt * 32 + base8];
      v16bf A = frag_from(ap, ap + 16);
      const unsigned short* wp = WFp + (((wave * KTILES + kt) << 9) + lane * 16);
      v16bf B = frag_from(wp, wp + 8);
      c = __builtin_amdgcn_wmma_f32_16x16x32_bf16(false, A, false, B, (short)0, c, false, false);
    }
    int col = wave * 16 + m;
    if (col < NCLS) {
      float bias = bfc[col];
      #pragma unroll
      for (int j = 0; j < 8; ++j) {
        int ro = blockIdx.x * 16 + j + base8;
        out[(size_t)ro * NCLS + col] = c[j] + bias;
      }
    }
  }
}

// ---------------------------------------------------------------------------
extern "C" void kernel_launch(void* const* d_in, const int* in_sizes, int n_in,
                              void* d_out, int out_size, void* d_ws, size_t ws_size,
                              hipStream_t stream) {
  (void)in_sizes; (void)n_in; (void)out_size; (void)ws_size;

  const int*   ids    = (const int*)  d_in[0];
  const int*   ids_l1 = (const int*)  d_in[1];
  const int*   ids_l2 = (const int*)  d_in[2];
  const float* feats  = (const float*)d_in[3];
  const float* wx1    = (const float*)d_in[4];
  const float* wn1    = (const float*)d_in[5];
  const float* wx2    = (const float*)d_in[6];
  const float* wn2    = (const float*)d_in[7];
  const float* w_fc   = (const float*)d_in[8];
  const float* b_fc   = (const float*)d_in[9];

  const int B = 4096, S1 = 25, S2 = 10;
  const int M1 = B * S1;                       // 102400 l1 nodes

  // workspace carve-out (256B aligned)
  char* ws = (char*)d_ws;
  size_t off = 0;
  auto carve = [&](size_t bytes) -> void* {
    void* p = ws + off;
    off += (bytes + 255) & ~(size_t)255;
    return p;
  };
  float* a1 = (float*)carve((size_t)M1 * DIN * sizeof(float));   // ~105 MB
  float* a0 = (float*)carve((size_t)B  * DIN * sizeof(float));
  float* b0 = (float*)carve((size_t)B  * DIN * sizeof(float));
  unsigned short* W1p = (unsigned short*)carve(16 * KTILES * 512 * sizeof(unsigned short));
  unsigned short* W2p = (unsigned short*)carve(16 * KTILES * 512 * sizeof(unsigned short));
  unsigned short* WFp = (unsigned short*)carve(3  * KTILES * 512 * sizeof(unsigned short));

  // 1) pre-pack weights into WMMA B-fragment order (bf16)
  pack2_kernel<<<256, 256, 0, stream>>>(wx1, wn1, W1p, 16 * KTILES * 512);
  pack2_kernel<<<256, 256, 0, stream>>>(wx2, wn2, W2p, 16 * KTILES * 512);
  pack1_kernel<<<48, 256, 0, stream>>>(w_fc, WFp, NCLS, 3 * KTILES * 512);

  // 2) layer-1 on l1 nodes: a1 = relu([h1@wx1 | mean10(h2)@wn1])
  sage_layer_kernel<<<M1 / 16, 256, 0, stream>>>(feats, ids_l1, feats, ids_l2,
                                                 S2, 1.f / S2, W1p, a1);
  // 3) layer-1 on roots: a0 = relu([h0@wx1 | mean25(h1)@wn1])
  sage_layer_kernel<<<B / 16, 256, 0, stream>>>(feats, ids, feats, ids_l1,
                                                S1, 1.f / S1, W1p, a0);
  // 4) layer-2: b0 = relu([a0@wx2 | mean25(a1)@wn2])  (identity indexing)
  sage_layer_kernel<<<B / 16, 256, 0, stream>>>(a0, nullptr, a1, nullptr,
                                                S1, 1.f / S1, W2p, b0);
  // 5) normalize rows + classifier
  final_kernel<<<B / 16, 128, 0, stream>>>(b0, WFp, b_fc, (float*)d_out);
}
